// Compressor_31963146617050
// MI455X (gfx1250) — matrix operand order repaired and checked by reading, not compiled, not run on hardware
//
#include <hip/hip_runtime.h>

// ---------- types ----------
typedef __attribute__((ext_vector_type(2)))  float          v2f;
typedef __attribute__((ext_vector_type(4)))  float          v4f;
typedef __attribute__((ext_vector_type(8)))  float          v8f;
typedef __attribute__((ext_vector_type(2)))  int            v2i;
typedef __attribute__((ext_vector_type(4)))  int            v4i;
typedef __attribute__((ext_vector_type(16))) __bf16         v16bf;

union FragBF { v4i q[2]; v16bf v; };

#define B_   4
#define S_   4096
#define K_   4096      // DIM
#define N_   2048      // COFF*HD
#define M_   16384     // B*S
#define HD_  512
#define RD_  64

// ---------- CDNA5 async-to-LDS availability (proxy for asm support) ----------
#if defined(__has_builtin)
#  if __has_builtin(__builtin_amdgcn_global_load_async_to_lds_b128) && \
      __has_builtin(__builtin_amdgcn_s_wait_asynccnt)
#    define USE_ASYNC_LDS 1
#  endif
#endif
#ifndef USE_ASYNC_LDS
#  define USE_ASYNC_LDS 0
#endif

#if USE_ASYNC_LDS
// VGLOBAL async encoding: VDST = LDS byte-address VGPR, VADDR = 64-bit global
// address VGPR pair, SADDR = off (GV mode). Tracked by ASYNCcnt.
__device__ __forceinline__ void async_load_b128(const void* gptr, void* lds_ptr) {
    unsigned long long ga = (unsigned long long)gptr;
    unsigned int       la = (unsigned int)(unsigned long long)lds_ptr; // generic->LDS offset (addr[31:0])
    asm volatile("global_load_async_to_lds_b128 %0, %1, off"
                 :: "v"(la), "v"(ga) : "memory");
}
#endif

// ---------- bf16 truncation-split helpers ----------
// hi = x truncated to bf16; lo = x - hi_f (exact in f32), truncated to bf16.
// Neglected lo*lo term in the 3-product GEMM => ~2^-15 relative error.
__device__ __forceinline__ unsigned short bf16trunc(float f) {
    return (unsigned short)(__float_as_uint(f) >> 16);
}
__device__ __forceinline__ float bf16hi_f(float f) {
    return __uint_as_float(__float_as_uint(f) & 0xFFFF0000u);
}

// =====================================================================
// Kernel 1: W = [wkv_w | wgate_w] (K x N) f32  ->  WT (N x K) bf16 hi/lo
// =====================================================================
__global__ __launch_bounds__(256) void pack_w_kernel(
    const float* __restrict__ wkv, const float* __restrict__ wgate,
    unsigned short* __restrict__ WTh, unsigned short* __restrict__ WTl)
{
    __shared__ float tile[32][33];
    const int tx = threadIdx.x;           // 32
    const int ty = threadIdx.y;           // 8
    const int k0 = blockIdx.x * 32;       // 128 blocks
    const int n0 = blockIdx.y * 32;       // 64 blocks

    #pragma unroll
    for (int yy = 0; yy < 4; ++yy) {
        int kg = k0 + ty + yy * 8;
        int ng = n0 + tx;
        float v = (ng < 1024) ? wkv[(size_t)kg * 1024 + ng]
                              : wgate[(size_t)kg * 1024 + (ng - 1024)];
        tile[ty + yy * 8][tx] = v;
    }
    __syncthreads();
    #pragma unroll
    for (int yy = 0; yy < 4; ++yy) {
        int nl = ty + yy * 8;
        int kl = tx;
        float v = tile[kl][nl];
        float lof = v - bf16hi_f(v);
        size_t idx = (size_t)(n0 + nl) * K_ + (k0 + kl);
        WTh[idx] = bf16trunc(v);
        WTl[idx] = bf16trunc(lof);
    }
}

// =====================================================================
// Kernel 2: C (M x N) = X (M x K) @ W (K x N), split-bf16 WMMA,
// software-pipelined: A regs prefetch + async-to-LDS double-buffered B.
// block 256 thr = 8 waves (4M x 2N), wave tile 32x64, block tile 128x128
// =====================================================================
__global__ __launch_bounds__(256) void gemm_kernel(
    const float* __restrict__ X,
    const unsigned short* __restrict__ WTh,
    const unsigned short* __restrict__ WTl,
    float* __restrict__ C)
{
#if USE_ASYNC_LDS
#define NBUF 2
#else
#define NBUF 1
#endif
    __shared__ unsigned short lAh[128 * 32];
    __shared__ unsigned short lAl[128 * 32];
    __shared__ unsigned short lBh[NBUF][128 * 32];
    __shared__ unsigned short lBl[NBUF][128 * 32];

    const int tid = threadIdx.x;
    const int m0 = blockIdx.y * 128;
    const int n0 = blockIdx.x * 128;
    const int wave = tid >> 5;
    const int lane = tid & 31;
    const int lm = lane & 15;
    const int lh = lane >> 4;
    const int wm = (wave & 3) * 32;   // 0..96
    const int wn = (wave >> 2) * 64;  // 0,64

    const v8f vzero = {0.f, 0.f, 0.f, 0.f, 0.f, 0.f, 0.f, 0.f};
    v8f acc[2][4];
    #pragma unroll
    for (int i = 0; i < 2; ++i)
        #pragma unroll
        for (int j = 0; j < 4; ++j) acc[i][j] = vzero;

    v4f aReg[4];
#if !USE_ASYNC_LDS
    v4i bRegH[2], bRegL[2];
#endif

    // ---- prologue: start fetching tile kk = 0
    #pragma unroll
    for (int cc = 0; cc < 4; ++cc) {
        int c = tid + cc * 256, r = c >> 3, q = c & 7;
        aReg[cc] = *(const v4f*)(X + (size_t)(m0 + r) * K_ + q * 4);
    }
    #pragma unroll
    for (int cc = 0; cc < 2; ++cc) {
        int c = tid + cc * 256, r = c >> 2, q = c & 3;
        size_t g = (size_t)(n0 + r) * K_ + q * 8;
#if USE_ASYNC_LDS
        async_load_b128(WTh + g, &lBh[0][r * 32 + q * 8]);
        async_load_b128(WTl + g, &lBl[0][r * 32 + q * 8]);
#else
        bRegH[cc] = *(const v4i*)(WTh + g);
        bRegL[cc] = *(const v4i*)(WTl + g);
#endif
    }

    for (int kk = 0; kk < K_; kk += 32) {
#if USE_ASYNC_LDS
        const int cur = (kk >> 5) & 1;
        const int nxt = cur ^ 1;
#else
        const int cur = 0;
#endif
        const bool more = (kk + 32) < K_;

        __syncthreads();   // B1: everyone done computing previous tile

        // ---- commit current A tile to LDS (f32 -> bf16 hi/lo, v_perm pack)
        #pragma unroll
        for (int cc = 0; cc < 4; ++cc) {
            int c = tid + cc * 256, r = c >> 3, q = c & 7;
            unsigned int u0 = __float_as_uint(aReg[cc].x);
            unsigned int u1 = __float_as_uint(aReg[cc].y);
            unsigned int u2 = __float_as_uint(aReg[cc].z);
            unsigned int u3 = __float_as_uint(aReg[cc].w);
            float l0 = aReg[cc].x - __uint_as_float(u0 & 0xFFFF0000u);
            float l1 = aReg[cc].y - __uint_as_float(u1 & 0xFFFF0000u);
            float l2 = aReg[cc].z - __uint_as_float(u2 & 0xFFFF0000u);
            float l3 = aReg[cc].w - __uint_as_float(u3 & 0xFFFF0000u);
            v2i hv = { (int)__builtin_amdgcn_perm(u1, u0, 0x07060302u),
                       (int)__builtin_amdgcn_perm(u3, u2, 0x07060302u) };
            v2i lv = { (int)__builtin_amdgcn_perm(__float_as_uint(l1), __float_as_uint(l0), 0x07060302u),
                       (int)__builtin_amdgcn_perm(__float_as_uint(l3), __float_as_uint(l2), 0x07060302u) };
            *(v2i*)&lAh[r * 32 + q * 4] = hv;
            *(v2i*)&lAl[r * 32 + q * 4] = lv;
        }
#if !USE_ASYNC_LDS
        // ---- commit current B tile to LDS
        #pragma unroll
        for (int cc = 0; cc < 2; ++cc) {
            int c = tid + cc * 256, r = c >> 2, q = c & 3;
            *(v4i*)&lBh[0][r * 32 + q * 8] = bRegH[cc];
            *(v4i*)&lBl[0][r * 32 + q * 8] = bRegL[cc];
        }
#endif
        // ---- prefetch next tile (overlaps with WMMA below)
        if (more) {
            #pragma unroll
            for (int cc = 0; cc < 4; ++cc) {
                int c = tid + cc * 256, r = c >> 3, q = c & 7;
                aReg[cc] = *(const v4f*)(X + (size_t)(m0 + r) * K_ + (kk + 32) + q * 4);
            }
            #pragma unroll
            for (int cc = 0; cc < 2; ++cc) {
                int c = tid + cc * 256, r = c >> 2, q = c & 3;
                size_t g = (size_t)(n0 + r) * K_ + (kk + 32) + q * 8;
#if USE_ASYNC_LDS
                async_load_b128(WTh + g, &lBh[nxt][r * 32 + q * 8]);
                async_load_b128(WTl + g, &lBl[nxt][r * 32 + q * 8]);
#else
                bRegH[cc] = *(const v4i*)(WTh + g);
                bRegL[cc] = *(const v4i*)(WTl + g);
#endif
            }
        }
#if USE_ASYNC_LDS
        // current buffer's 4 async loads complete; next tile's 4 stay in flight
        if (more) __builtin_amdgcn_s_wait_asynccnt(4);
        else      __builtin_amdgcn_s_wait_asynccnt(0);
#endif
        __syncthreads();   // B2: LDS tile (ds stores + async B) visible to all

        // ---- fragments + WMMA
        FragBF ah[2], al[2];
        #pragma unroll
        for (int i = 0; i < 2; ++i) {
            int ar = (wm + i * 16 + lm) * 32;
            ah[i].q[0] = *(const v4i*)&lAh[ar + lh * 8];
            ah[i].q[1] = *(const v4i*)&lAh[ar + 16 + lh * 8];
            al[i].q[0] = *(const v4i*)&lAl[ar + lh * 8];
            al[i].q[1] = *(const v4i*)&lAl[ar + 16 + lh * 8];
        }
        #pragma unroll
        for (int j = 0; j < 4; ++j) {
            int br = (wn + j * 16 + lm) * 32 + lh * 8;
            FragBF bh, bl;
            bh.q[0] = *(const v4i*)&lBh[cur][br];
            bh.q[1] = *(const v4i*)&lBh[cur][br + 16];
            bl.q[0] = *(const v4i*)&lBl[cur][br];
            bl.q[1] = *(const v4i*)&lBl[cur][br + 16];
            #pragma unroll
            for (int i = 0; i < 2; ++i) {
                acc[i][j] = __builtin_amdgcn_wmma_f32_16x16x32_bf16(
                    false, ah[i].v, false, bh.v, (short)0, acc[i][j], false, false);
                acc[i][j] = __builtin_amdgcn_wmma_f32_16x16x32_bf16(
                    false, ah[i].v, false, bl.v, (short)0, acc[i][j], false, false);
                acc[i][j] = __builtin_amdgcn_wmma_f32_16x16x32_bf16(
                    false, al[i].v, false, bh.v, (short)0, acc[i][j], false, false);
            }
        }
    }

    // ---- epilogue: C layout: VGPR v -> row v (+8 for hi half-wave), col = lm
    #pragma unroll
    for (int i = 0; i < 2; ++i) {
        #pragma unroll
        for (int j = 0; j < 4; ++j) {
            int gm = m0 + wm + i * 16 + lh * 8;
            int gn = n0 + wn + j * 16 + lm;
            #pragma unroll
            for (int v = 0; v < 8; ++v)
                C[(size_t)(gm + v) * N_ + gn] = acc[i][j][v];
        }
    }
}

// =====================================================================
// Kernel 3: window softmax + fold + RMSNorm + RoPE + cache scatter
// one block per (b,t); thread owns folded cols 2*tid, 2*tid+1
// =====================================================================
__global__ __launch_bounds__(256) void stage2_kernel(
    const float* __restrict__ C,
    const float* __restrict__ ape,      // (4,1024)
    const float* __restrict__ normw,    // (512)
    const float* __restrict__ cosT,     // (4096,32)
    const float* __restrict__ sinT,
    const int*   __restrict__ blkoff,   // (64)
    float* __restrict__ out)            // ckv_flat (16384*512) ++ cache (4096*512)
{
    __shared__ float red[256];
    const int bt = blockIdx.x;
    const int b = bt >> 12;
    const int t = bt & (S_ - 1);
    const int tid = threadIdx.x;
    const int j0 = tid * 2;

    float kvals[4][4], svals[4][4];
    int   apr[4];
    bool  vld[4];
    #pragma unroll
    for (int r = 0; r < 4; ++r) {
        int idx = t - 3 + r;
        vld[r] = (idx >= 0);
        int idxc = idx > 0 ? idx : 0;
        apr[r] = idxc & 3;
        const float* rowp = C + (size_t)(b * S_ + idxc) * N_;
        v2f a  = *(const v2f*)(rowp + j0);            // kv cols j0, j0+1
        v2f a2 = *(const v2f*)(rowp + 512 + j0);      // kv cols j0+512, +513
        v2f s  = *(const v2f*)(rowp + 1024 + j0);     // sc cols j0, j0+1
        v2f s2 = *(const v2f*)(rowp + 1536 + j0);     // sc cols j0+512, +513
        kvals[r][0] = a.x;  kvals[r][1] = a.y;
        kvals[r][2] = a2.x; kvals[r][3] = a2.y;
        svals[r][0] = s.x;  svals[r][1] = s.y;
        svals[r][2] = s2.x; svals[r][3] = s2.y;
    }

    float o[4];
    #pragma unroll
    for (int c = 0; c < 4; ++c) {
        int jc = (c < 2) ? (j0 + c) : (j0 + 512 + (c - 2));
        float mx = -3.4e38f;
        #pragma unroll
        for (int r = 0; r < 4; ++r)
            if (vld[r]) mx = fmaxf(mx, svals[r][c]);
        float sum = 0.f, accv = 0.f;
        #pragma unroll
        for (int r = 0; r < 4; ++r) {
            if (vld[r]) {
                float w = expf(svals[r][c] - mx);
                sum += w;
                accv += w * (kvals[r][c] + ape[apr[r] * 1024 + jc]);
            }
        }
        o[c] = accv / sum;
    }
    float f0 = o[0] + o[2];
    float f1 = o[1] + o[3];

    // block reduction of sum of squares over 512 folded values
    red[tid] = f0 * f0 + f1 * f1;
    __syncthreads();
    for (int s = 128; s > 0; s >>= 1) {
        if (tid < s) red[tid] += red[tid + s];
        __syncthreads();
    }
    float scale = rsqrtf(red[0] * (1.0f / 512.0f) + 1e-6f);

    float o0 = f0 * scale * normw[j0];
    float o1 = f1 * scale * normw[j0 + 1];

    // RoPE on last 64 columns: tid >= 224 owns pair k = tid-224
    if (tid >= 224) {
        int k = tid - 224;
        float cc = cosT[t * 32 + k];
        float ss = sinT[t * 32 + k];
        float r0 = o0 * cc - o1 * ss;
        float r1 = o0 * ss + o1 * cc;
        o0 = r0; o1 = r1;
    }

    v2f res = {o0, o1};
    *(v2f*)(out + (size_t)bt * HD_ + j0) = res;

    if ((t & 3) == 3) {
        int cidx = t >> 2;                            // 0..1023
        int blk = blkoff[b * 16 + (cidx >> 6)];
        int row = blk * 64 + (cidx & 63);
        float* cache = out + (size_t)M_ * HD_;
        *(v2f*)(cache + (size_t)row * HD_ + j0) = res;
    }
}

// =====================================================================
extern "C" void kernel_launch(void* const* d_in, const int* in_sizes, int n_in,
                              void* d_out, int out_size, void* d_ws, size_t ws_size,
                              hipStream_t stream) {
    const float* x      = (const float*)d_in[0];
    const float* wkv    = (const float*)d_in[1];
    const float* wgate  = (const float*)d_in[2];
    const float* ape    = (const float*)d_in[3];
    const float* normw  = (const float*)d_in[4];
    const float* cosT   = (const float*)d_in[5];
    const float* sinT   = (const float*)d_in[6];
    const int*   blkoff = (const int*)d_in[7];
    float* out = (float*)d_out;

    // workspace: WTh (16 MiB) | WTl (16 MiB) | C (128 MiB)
    unsigned short* WTh = (unsigned short*)d_ws;
    unsigned short* WTl = WTh + (size_t)N_ * K_;
    float* C = (float*)(WTl + (size_t)N_ * K_);

    pack_w_kernel<<<dim3(K_ / 32, N_ / 32), dim3(32, 8), 0, stream>>>(wkv, wgate, WTh, WTl);
    gemm_kernel<<<dim3(N_ / 128, M_ / 128), 256, 0, stream>>>(x, WTh, WTl, C);
    stage2_kernel<<<M_, 256, 0, stream>>>(C, ape, normw, cosT, sinT, blkoff, out);
}
